// DeformableIAMDouble_30846455120526
// MI455X (gfx1250) — matrix operand order repaired and checked by reading, not compiled, not run on hardware
//
#include <hip/hip_runtime.h>
#include <math.h>

#define B_   8
#define CI_  256
#define CO_  256
#define H_   64
#define W_   64
#define HW_  (H_ * W_)
#define NE_  (B_ * CO_ * H_ * W_)   // 8388608 elements

typedef __attribute__((ext_vector_type(16))) __bf16 v16bf;
typedef __attribute__((ext_vector_type(8)))  float  v8f;
typedef unsigned int u32x4 __attribute__((ext_vector_type(4)));
typedef int          i32x8 __attribute__((ext_vector_type(8)));
typedef int          i32x4 __attribute__((ext_vector_type(4)));

union Frag16 { unsigned int u[8]; uint4 u4[2]; v16bf v; };

// bilinear taps: 4 weights + 4 precomputed flat offsets (y*W+x, clamped)
struct alignas(16) PrepT { float w00, w01, w10, w11; int o00, o01, o10, o11; };

__device__ __forceinline__ unsigned int f2bf(float f) {
    unsigned int u = __float_as_uint(f);
    unsigned int r = u + 0x7FFFu + ((u >> 16) & 1u);   // round-to-nearest-even
    return r >> 16;
}
__device__ __forceinline__ unsigned int pack_bf2(float lo, float hi) {
    return f2bf(lo) | (f2bf(hi) << 16);
}
__device__ __forceinline__ int iclamp(int v, int lo, int hi) {
    return v < lo ? lo : (v > hi ? hi : v);
}
// storage permutation so each lane's 8 A-fragment dwords are contiguous
__device__ __forceinline__ int apermu(int j) {
    return (j & 3) | ((j & 8) >> 1) | ((j & 4) << 1);
}
__device__ __forceinline__ v8f wmma_bf16(const Frag16& a, const Frag16& b, v8f c) {
    return __builtin_amdgcn_wmma_f32_16x16x32_bf16(false, a.v, false, b.v,
                                                   (short)0, c, false, false);
}

// LDS tile geometry: row stride 20 dwords (80B: 16B aligned, bank-spread)
#define ASTRIDE 20
#define ABUF    (64 * ASTRIDE)     // 1280 dwords per buffer
#define BBUF    (128 * ASTRIDE)    // 2560 dwords per buffer

// ---------------------------------------------------------------------------
// TDM descriptor load: copy one 2048-dword weight chunk global -> LDS with
// the Tensor Data Mover.  pad_interval=16 dwords / pad_amount=4 dwords
// reproduces the ASTRIDE-20 row layout (16 data + 4 pad per N row).
// Issued by one wave; completion via s_wait_tensorcnt.
// (6-arg builtin form: g0, g1, g2, g3, g-extra, cpol)
// ---------------------------------------------------------------------------
__device__ __forceinline__ void tdm_load_chunk(const unsigned int* gsrc, unsigned int* ldsdst) {
    unsigned long long ga = (unsigned long long)(size_t)gsrc;
    unsigned lds = (unsigned)(size_t)ldsdst;   // low 32 bits = LDS byte address
    u32x4 g0;
    g0.x = 1u;                                  // count=1 (valid user descriptor)
    g0.y = lds;                                 // lds_addr
    g0.z = (unsigned)ga;                        // global_addr[31:0]
    g0.w = (unsigned)(ga >> 32) | (2u << 30);   // global_addr[56:32] | type=2
    i32x8 g1;
    g1[0] = (2 << 16)      // data_size = 4 bytes
          | (1 << 20)      // pad_enable
          | (3 << 22)      // pad_interval: 16 dwords
          | (3 << 25);     // pad_amount: 4 dwords
    g1[1] = (2048 << 16);  // tensor_dim0[15:0] = 2048 (bits 63:48)
    g1[2] = (1 << 16);     // tensor_dim0[31:16]=0, tensor_dim1 = 1
    g1[3] = 2048;          // tile_dim0 = 2048 dwords
    g1[4] = 1;             // tile_dim1 = 1
    g1[5] = 2048;          // tensor_dim0_stride[31:0]
    g1[6] = 0;
    g1[7] = 0;
    i32x4 z4 = {0, 0, 0, 0};
    i32x8 z8 = {0, 0, 0, 0, 0, 0, 0, 0};
    __builtin_amdgcn_tensor_load_to_lds(g0, g1, z4, z4, z8, 0);
}

// ---------------------------------------------------------------------------
// Weight prepack into tap-major/channel-inner K-order (k = p*Cin + c), bf16
// pairs in LDS-tile chunk layout.  Chunk (nt,kstep) = 2048 dwords,
// local = nl*16 + j, n = nt*128 + nl.
// ---------------------------------------------------------------------------
__global__ void prepack_w_kernel(const float* __restrict__ w, unsigned int* __restrict__ wpack,
                                 int Cin, int KK, int ntiles) {
    int Kdim   = Cin * KK;
    int ksteps = Kdim / 32;
    int d = blockIdx.x * 256 + threadIdx.x;
    int total = ntiles * ksteps * 2048;
    if (d >= total) return;
    int chunk = d >> 11;
    int local = d & 2047;
    int nl = local >> 4, j = local & 15;
    int nt = chunk / ksteps, kst = chunk - nt * ksteps;
    int n  = nt * 128 + nl;
    int k0 = kst * 32 + 2 * apermu(j);
    float v[2];
    #pragma unroll
    for (int e = 0; e < 2; ++e) {
        int kidx = k0 + e;
        int p = kidx / Cin;            // tap index
        int c = kidx - p * Cin;        // channel
        v[e] = w[(size_t)n * Kdim + c * KK + p];
    }
    wpack[d] = pack_bf2(v[0], v[1]);
}

// ---------------------------------------------------------------------------
// Wide implicit-GEMM conv: bf16 WMMA, 2x2 register tiling per wave
// (8 waves = M64 x N128 block tile), double-buffered LDS, prepacked weights,
// TDM-staged B tiles.  K-order: k = p*256 + c.
// ---------------------------------------------------------------------------
template <int KS, int PAD>
__global__ __launch_bounds__(256)
void conv_wmma128_kernel(const float* __restrict__ x, const unsigned int* __restrict__ wpack,
                         const float* __restrict__ bias, float* __restrict__ y,
                         int Cin, int Cout) {
    const int KK     = KS * KS;
    const int Kdim   = Cin * KK;
    const int ksteps = Kdim / 32;
    __shared__ unsigned int As[2 * ABUF];
    __shared__ unsigned int Bs[2 * BBUF];

    const int tid   = threadIdx.x;
    const int lane  = tid & 31;
    const int wid   = tid >> 5;
    const int lrow  = lane & 15;
    const int hi16  = lane >> 4;
    const int mt    = blockIdx.x;          // = b*H + h
    const int b     = mt / H_;
    const int h     = mt % H_;
    const int nbase = blockIdx.y * 128;
    const int mw    = (wid & 1) * 32;
    const int nw    = (wid >> 1) * 32;
    const unsigned int* wp = wpack + (size_t)blockIdx.y * ksteps * 2048;

    auto stageA = [&](int kst, int buf) {
        const int p     = kst >> 3;            // tap (0 for 1x1)
        const int ky    = p / 3;
        const int kx    = p - ky * 3;
        const int ih    = h + ky - PAD;
        const bool vy   = (ih >= 0) && (ih < H_);
        const int cbase = (kst & 7) * 32;
        #pragma unroll
        for (int t = 0; t < 4; ++t) {
            int slot = tid + t * 256;          // 1024 dword slots
            int j    = slot >> 6;              // K-pair index (c pair)
            int row  = slot & 63;              // output w coordinate
            int c    = cbase + 2 * j;
            int iw   = row + kx - PAD;
            bool ok  = vy && (iw >= 0) && (iw < W_);
            const float* px = x + ((size_t)(b * Cin + c) * H_ + ih) * W_ + iw;
            float v0 = ok ? px[0]   : 0.f;     // channel c
            float v1 = ok ? px[HW_] : 0.f;     // channel c+1, same (ih,iw)
            As[buf * ABUF + row * ASTRIDE + apermu(j)] = pack_bf2(v0, v1);
        }
    };
    auto stageB = [&](int kst, int buf) {
        if (wid == 0)
            tdm_load_chunk(wp + (size_t)kst * 2048, &Bs[buf * BBUF]);
    };

    v8f acc00 = {}, acc01 = {}, acc10 = {}, acc11 = {};

    stageA(0, 0);
    stageB(0, 0);
    if (wid == 0) __builtin_amdgcn_s_wait_tensorcnt(0);
    __syncthreads();
    for (int ks = 0; ks < ksteps; ++ks) {
        const int cur = ks & 1;
        if (ks + 1 < ksteps) { stageA(ks + 1, cur ^ 1); stageB(ks + 1, cur ^ 1); }
        if (ks + 2 < ksteps)
            __builtin_prefetch(wp + (size_t)(ks + 2) * 2048 + tid * 8, 0, 0);

        Frag16 fa0, fa1, fb0, fb1;
        {
            const uint4* p0 = (const uint4*)&As[cur * ABUF + (mw + lrow) * ASTRIDE + hi16 * 8];
            const uint4* p1 = (const uint4*)&As[cur * ABUF + (mw + 16 + lrow) * ASTRIDE + hi16 * 8];
            fa0.u4[0] = p0[0]; fa0.u4[1] = p0[1];
            fa1.u4[0] = p1[0]; fa1.u4[1] = p1[1];
            const uint4* q0 = (const uint4*)&Bs[cur * BBUF + (nw + lrow) * ASTRIDE + hi16 * 8];
            const uint4* q1 = (const uint4*)&Bs[cur * BBUF + (nw + 16 + lrow) * ASTRIDE + hi16 * 8];
            fb0.u4[0] = q0[0]; fb0.u4[1] = q0[1];
            fb1.u4[0] = q1[0]; fb1.u4[1] = q1[1];
        }
        acc00 = wmma_bf16(fa0, fb0, acc00);
        acc01 = wmma_bf16(fa0, fb1, acc01);
        acc10 = wmma_bf16(fa1, fb0, acc10);
        acc11 = wmma_bf16(fa1, fb1, acc11);
        if (wid == 0) __builtin_amdgcn_s_wait_tensorcnt(0);
        __syncthreads();
    }

    const int n0 = nbase + nw + lrow;
    const int n1 = n0 + 16;
    const float bv0 = bias[n0];
    const float bv1 = bias[n1];
    #pragma unroll
    for (int r = 0; r < 8; ++r) {
        int wp0 = mw + r + (hi16 ? 8 : 0);
        int wp1 = wp0 + 16;
        y[((b * Cout + n0) * H_ + h) * W_ + wp0] = acc00[r] + bv0;
        y[((b * Cout + n1) * H_ + h) * W_ + wp0] = acc01[r] + bv1;
        y[((b * Cout + n0) * H_ + h) * W_ + wp1] = acc10[r] + bv0;
        y[((b * Cout + n1) * H_ + h) * W_ + wp1] = acc11[r] + bv1;
    }
}

// ---------------------------------------------------------------------------
// Narrow (N=32) conv for the 32-channel offset+modulation head.
// (Self-consistent old K-order; reads raw f32 wcat weights.)
// ---------------------------------------------------------------------------
template <int KS, int PAD>
__global__ __launch_bounds__(256)
void conv_wmma32_kernel(const float* __restrict__ x, const float* __restrict__ w,
                        const float* __restrict__ bias, float* __restrict__ y,
                        int Cin, int Cout) {
    const int KK   = KS * KS;
    const int Kdim = Cin * KK;
    __shared__ unsigned int As[64 * 17];
    __shared__ unsigned int Bs[32 * 17];

    const int tid   = threadIdx.x;
    const int lane  = tid & 31;
    const int wid   = tid >> 5;
    const int lrow  = lane & 15;
    const int hi16  = lane >> 4;
    const int mt    = blockIdx.x;
    const int b     = mt / H_;
    const int h     = mt % H_;
    const int m_sub = wid & 3;
    const int n_sub = wid >> 2;

    v8f acc = {};

    for (int kk0 = 0; kk0 < Kdim; kk0 += 32) {
        __syncthreads();
        #pragma unroll
        for (int t = 0; t < 4; ++t) {
            int slot = tid + t * 256;
            int row  = slot >> 4;
            int j    = slot & 15;
            float vv[2];
            #pragma unroll
            for (int e = 0; e < 2; ++e) {
                int k  = kk0 + 2 * j + e;
                int c  = k / KK;
                int r  = k - c * KK;
                int ky = r / KS;
                int kx = r - ky * KS;
                int ih = h + ky - PAD;
                int iw = row + kx - PAD;
                vv[e] = (ih >= 0 && ih < H_ && iw >= 0 && iw < W_)
                            ? x[((b * Cin + c) * H_ + ih) * W_ + iw] : 0.f;
            }
            As[row * 17 + j] = pack_bf2(vv[0], vv[1]);
        }
        #pragma unroll
        for (int t = 0; t < 2; ++t) {
            int slot = tid + t * 256;
            int nl   = slot >> 4;
            int j    = slot & 15;
            const float2 wv = *(const float2*)(w + (size_t)nl * Kdim + kk0 + 2 * j);
            Bs[nl * 17 + j] = pack_bf2(wv.x, wv.y);
        }
        __syncthreads();
        Frag16 fa, fb;
        #pragma unroll
        for (int v = 0; v < 8; ++v) {
            int di = (v & 3) + ((v & 4) ? 8 : 0) + (hi16 ? 4 : 0);
            fa.u[v] = As[(m_sub * 16 + lrow) * 17 + di];
            fb.u[v] = Bs[(n_sub * 16 + lrow) * 17 + di];
        }
        acc = wmma_bf16(fa, fb, acc);
    }

    const int n  = n_sub * 16 + lrow;
    const float bv = bias[n];
    #pragma unroll
    for (int r = 0; r < 8; ++r) {
        int wp = m_sub * 16 + r + (hi16 ? 8 : 0);
        y[((b * Cout + n) * H_ + h) * W_ + wp] = acc[r] + bv;
    }
}

// ---------------------------------------------------------------------------
// Modulated deformable conv GEMM: 2x2 tiling, double-buffered LDS,
// TDM-staged prepacked weights (tap-major K-order), prep table in LDS.
// ---------------------------------------------------------------------------
__global__ __launch_bounds__(256)
void deform_wmma_kernel(const float* __restrict__ src, const PrepT* __restrict__ prep,
                        const unsigned int* __restrict__ wpack, const float* __restrict__ bias,
                        float* __restrict__ y) {
    const int ksteps = (CI_ * 9) / 32;
    __shared__ unsigned int As[2 * ABUF];
    __shared__ unsigned int Bs[2 * BBUF];
    __shared__ PrepT sPrep[64 * 9];     // 18KB

    const int tid   = threadIdx.x;
    const int lane  = tid & 31;
    const int wid   = tid >> 5;
    const int lrow  = lane & 15;
    const int hi16  = lane >> 4;
    const int mt    = blockIdx.x;
    const int b     = mt / H_;
    const int h     = mt % H_;
    const int nbase = blockIdx.y * 128;
    const int mw    = (wid & 1) * 32;
    const int nw    = (wid >> 1) * 32;
    const unsigned int* wp = wpack + (size_t)blockIdx.y * ksteps * 2048;

    // cache this block's prep entries: sPrep[row*9+p]
    for (int i = tid; i < 64 * 9; i += 256) {
        int row = i / 9, p = i - row * 9;
        sPrep[i] = prep[((b * 9 + p) * H_ + h) * W_ + row];
    }

    auto stageA = [&](int kst, int buf) {
        const int p     = kst >> 3;
        const int cbase = (kst & 7) * 32;
        #pragma unroll
        for (int t = 0; t < 4; ++t) {
            int slot = tid + t * 256;
            int j    = slot >> 6;
            int row  = slot & 63;
            int c    = cbase + 2 * j;
            PrepT pr = sPrep[row * 9 + p];
            const float* base = src + (size_t)(b * CI_ + c) * HW_;
            float v0 = pr.w00 * base[pr.o00] + pr.w01 * base[pr.o01]
                     + pr.w10 * base[pr.o10] + pr.w11 * base[pr.o11];
            const float* base1 = base + HW_;
            float v1 = pr.w00 * base1[pr.o00] + pr.w01 * base1[pr.o01]
                     + pr.w10 * base1[pr.o10] + pr.w11 * base1[pr.o11];
            As[buf * ABUF + row * ASTRIDE + apermu(j)] = pack_bf2(v0, v1);
        }
    };
    auto stageB = [&](int kst, int buf) {
        if (wid == 0)
            tdm_load_chunk(wp + (size_t)kst * 2048, &Bs[buf * BBUF]);
    };

    v8f acc00 = {}, acc01 = {}, acc10 = {}, acc11 = {};

    __syncthreads();        // sPrep ready
    stageA(0, 0);
    stageB(0, 0);
    if (wid == 0) __builtin_amdgcn_s_wait_tensorcnt(0);
    __syncthreads();
    for (int ks = 0; ks < ksteps; ++ks) {
        const int cur = ks & 1;
        if (ks + 1 < ksteps) { stageA(ks + 1, cur ^ 1); stageB(ks + 1, cur ^ 1); }
        if (ks + 2 < ksteps)
            __builtin_prefetch(wp + (size_t)(ks + 2) * 2048 + tid * 8, 0, 0);

        Frag16 fa0, fa1, fb0, fb1;
        {
            const uint4* p0 = (const uint4*)&As[cur * ABUF + (mw + lrow) * ASTRIDE + hi16 * 8];
            const uint4* p1 = (const uint4*)&As[cur * ABUF + (mw + 16 + lrow) * ASTRIDE + hi16 * 8];
            fa0.u4[0] = p0[0]; fa0.u4[1] = p0[1];
            fa1.u4[0] = p1[0]; fa1.u4[1] = p1[1];
            const uint4* q0 = (const uint4*)&Bs[cur * BBUF + (nw + lrow) * ASTRIDE + hi16 * 8];
            const uint4* q1 = (const uint4*)&Bs[cur * BBUF + (nw + 16 + lrow) * ASTRIDE + hi16 * 8];
            fb0.u4[0] = q0[0]; fb0.u4[1] = q0[1];
            fb1.u4[0] = q1[0]; fb1.u4[1] = q1[1];
        }
        acc00 = wmma_bf16(fa0, fb0, acc00);
        acc01 = wmma_bf16(fa0, fb1, acc01);
        acc10 = wmma_bf16(fa1, fb0, acc10);
        acc11 = wmma_bf16(fa1, fb1, acc11);
        if (wid == 0) __builtin_amdgcn_s_wait_tensorcnt(0);
        __syncthreads();
    }

    const int n0 = nbase + nw + lrow;
    const int n1 = n0 + 16;
    const float bv0 = bias[n0];
    const float bv1 = bias[n1];
    #pragma unroll
    for (int r = 0; r < 8; ++r) {
        int wp0 = mw + r + (hi16 ? 8 : 0);
        int wp1 = wp0 + 16;
        y[((b * CO_ + n0) * H_ + h) * W_ + wp0] = acc00[r] + bv0;
        y[((b * CO_ + n1) * H_ + h) * W_ + wp0] = acc01[r] + bv1;
        y[((b * CO_ + n0) * H_ + h) * W_ + wp1] = acc10[r] + bv0;
        y[((b * CO_ + n1) * H_ + h) * W_ + wp1] = acc11[r] + bv1;
    }
}

// ---------------------------------------------------------------------------
// Bilinear sampling prep per (b,tap,h,w).
// ---------------------------------------------------------------------------
__global__ void deform_prep_kernel(const float* __restrict__ offmod, PrepT* __restrict__ prep) {
    int gid = blockIdx.x * blockDim.x + threadIdx.x;
    if (gid >= B_ * 9 * HW_) return;
    int b   = gid / (9 * HW_);
    int rem = gid - b * (9 * HW_);
    int p   = rem / HW_;
    int hw  = rem - p * HW_;
    int h   = hw / W_;
    int w   = hw - h * W_;

    const float* om = offmod + (size_t)b * 32 * HW_;
    float dy = om[(2 * p) * HW_ + hw];
    float dx = om[(2 * p + 1) * HW_ + hw];
    float mr = om[(18 + p) * HW_ + hw];
    float m  = 2.f / (1.f + __expf(-mr));

    float py = (float)(h - 1 + p / 3) + dy;
    float px = (float)(w - 1 + p % 3) + dx;
    float y0f = floorf(py), x0f = floorf(px);
    float ly = py - y0f, lx = px - x0f;
    int y0 = (int)y0f, x0 = (int)x0f;
    int y1 = y0 + 1,   x1 = x0 + 1;
    float vy0 = (y0 >= 0 && y0 < H_) ? 1.f : 0.f;
    float vy1 = (y1 >= 0 && y1 < H_) ? 1.f : 0.f;
    float vx0 = (x0 >= 0 && x0 < W_) ? 1.f : 0.f;
    float vx1 = (x1 >= 0 && x1 < W_) ? 1.f : 0.f;

    int y0c = iclamp(y0, 0, H_ - 1), y1c = iclamp(y1, 0, H_ - 1);
    int x0c = iclamp(x0, 0, W_ - 1), x1c = iclamp(x1, 0, W_ - 1);

    PrepT pr;
    pr.w00 = (1.f - ly) * (1.f - lx) * vy0 * vx0 * m;
    pr.w01 = (1.f - ly) * lx         * vy0 * vx1 * m;
    pr.w10 = ly         * (1.f - lx) * vy1 * vx0 * m;
    pr.w11 = ly         * lx         * vy1 * vx1 * m;
    pr.o00 = y0c * W_ + x0c;  pr.o01 = y0c * W_ + x1c;
    pr.o10 = y1c * W_ + x0c;  pr.o11 = y1c * W_ + x1c;
    prep[gid] = pr;
}

// ---------------------------------------------------------------------------
// Per-channel BN statistics; one block per channel, float4 loads.
// ---------------------------------------------------------------------------
__global__ __launch_bounds__(256)
void bn_stats_kernel(const float* __restrict__ x, float* __restrict__ mean,
                     float* __restrict__ var) {
    const int c = blockIdx.x;
    float s = 0.f, s2 = 0.f;
    for (int i = threadIdx.x; i < (B_ * HW_) / 4; i += 256) {
        int b   = i / (HW_ / 4);
        int hw4 = i - b * (HW_ / 4);
        float4 v = ((const float4*)(x + ((size_t)b * CO_ + c) * HW_))[hw4];
        s  += v.x + v.y + v.z + v.w;
        s2 += v.x * v.x + v.y * v.y + v.z * v.z + v.w * v.w;
    }
    __shared__ float sh[256], sh2[256];
    sh[threadIdx.x] = s; sh2[threadIdx.x] = s2;
    __syncthreads();
    for (int st = 128; st > 0; st >>= 1) {
        if (threadIdx.x < st) {
            sh[threadIdx.x]  += sh[threadIdx.x + st];
            sh2[threadIdx.x] += sh2[threadIdx.x + st];
        }
        __syncthreads();
    }
    if (threadIdx.x == 0) {
        const float inv = 1.f / (float)(B_ * HW_);
        float mn = sh[0] * inv;
        mean[c] = mn;
        var[c]  = sh2[0] * inv - mn * mn;
    }
}

__global__ void bn_relu_kernel(const float* __restrict__ x, const float* __restrict__ mean,
                               const float* __restrict__ var, const float* __restrict__ g,
                               const float* __restrict__ bt, float* __restrict__ y) {
    size_t i4 = (size_t)blockIdx.x * blockDim.x + threadIdx.x;   // over NE_/4
    int c = (int)((i4 * 4 / HW_) % CO_);
    float sc = g[c] * rsqrtf(var[c] + 1e-5f);
    float mn = mean[c], bb = bt[c];
    float4 v = ((const float4*)x)[i4];
    float4 o;
    o.x = (v.x - mn) * sc + bb; o.x = o.x > 0.f ? o.x : 0.f;
    o.y = (v.y - mn) * sc + bb; o.y = o.y > 0.f ? o.y : 0.f;
    o.z = (v.z - mn) * sc + bb; o.z = o.z > 0.f ? o.z : 0.f;
    o.w = (v.w - mn) * sc + bb; o.w = o.w > 0.f ? o.w : 0.f;
    ((float4*)y)[i4] = o;
}

__global__ void final_fuse_kernel(const float* __restrict__ a, const float* __restrict__ m2,
                                  const float* __restrict__ v2, const float* __restrict__ g2,
                                  const float* __restrict__ bt2, const float* __restrict__ r,
                                  const float* __restrict__ m3, const float* __restrict__ v3,
                                  const float* __restrict__ g3, const float* __restrict__ bt3,
                                  float* __restrict__ out) {
    size_t i4 = (size_t)blockIdx.x * blockDim.x + threadIdx.x;   // over NE_/4
    int c = (int)((i4 * 4 / HW_) % CO_);
    float sa = g2[c] * rsqrtf(v2[c] + 1e-5f), ma = m2[c], ba = bt2[c];
    float sr = g3[c] * rsqrtf(v3[c] + 1e-5f), mr = m3[c], br = bt3[c];
    float4 va = ((const float4*)a)[i4];
    float4 vr = ((const float4*)r)[i4];
    float4 o;
    o.x = (va.x - ma) * sa + ba + (vr.x - mr) * sr + br; o.x = o.x > 0.f ? o.x : 0.f;
    o.y = (va.y - ma) * sa + ba + (vr.y - mr) * sr + br; o.y = o.y > 0.f ? o.y : 0.f;
    o.z = (va.z - ma) * sa + ba + (vr.z - mr) * sr + br; o.z = o.z > 0.f ? o.z : 0.f;
    o.w = (va.w - ma) * sa + ba + (vr.w - mr) * sr + br; o.w = o.w > 0.f ? o.w : 0.f;
    ((float4*)out)[i4] = o;
}

// Concatenate offset+modulation conv weights into one zero-padded 32-channel conv.
__global__ void build_wcat_kernel(const float* __restrict__ w_off, const float* __restrict__ b_off,
                                  const float* __restrict__ w_mod, const float* __restrict__ b_mod,
                                  float* __restrict__ wcat, float* __restrict__ bcat) {
    int i = blockIdx.x * blockDim.x + threadIdx.x;
    if (i < 32)
        bcat[i] = (i < 18) ? b_off[i] : (i < 27 ? b_mod[i - 18] : 0.f);
    if (i >= 32 * 2304) return;
    int o  = i / 2304;
    int rr = i - o * 2304;
    wcat[i] = (o < 18) ? w_off[o * 2304 + rr]
                       : (o < 27 ? w_mod[(o - 18) * 2304 + rr] : 0.f);
}

extern "C" void kernel_launch(void* const* d_in, const int* in_sizes, int n_in,
                              void* d_out, int out_size, void* d_ws, size_t ws_size,
                              hipStream_t stream) {
    const float* x     = (const float*)d_in[0];
    const float* w1    = (const float*)d_in[2];
    const float* b1    = (const float*)d_in[3];
    const float* g1    = (const float*)d_in[4];
    const float* bt1   = (const float*)d_in[5];
    const float* w_off = (const float*)d_in[6];
    const float* b_off = (const float*)d_in[7];
    const float* w_mod = (const float*)d_in[8];
    const float* b_mod = (const float*)d_in[9];
    const float* w_d   = (const float*)d_in[10];
    const float* b_d   = (const float*)d_in[11];
    const float* g2    = (const float*)d_in[12];
    const float* bt2   = (const float*)d_in[13];
    const float* w_ds  = (const float*)d_in[14];
    const float* b_ds  = (const float*)d_in[15];
    const float* g3    = (const float*)d_in[16];
    const float* bt3   = (const float*)d_in[17];

    float* ws     = (float*)d_ws;
    float* rawA   = ws;                                    // conv1 raw, later deform raw
    float* out1   = rawA + (size_t)NE_;
    float* dsraw  = out1 + (size_t)NE_;
    float* offmod = dsraw + (size_t)NE_;                   // [B,32,H,W]
    float* prepf  = offmod + (size_t)B_ * 32 * HW_;        // PrepT[B*9*H*W]
    PrepT* prep   = (PrepT*)prepf;
    float* wcat   = prepf + (size_t)B_ * 9 * HW_ * 8;      // [32,256,3,3]
    float* bcat   = wcat + 32 * 2304;
    float* stats  = bcat + 32;                             // 6 x 256
    float *mean1 = stats,        *var1 = stats + 256;
    float *mean2 = stats + 512,  *var2 = stats + 768;
    float *mean3 = stats + 1024, *var3 = stats + 1280;
    unsigned int* wpack1  = (unsigned int*)(stats + 1536);      // 2*72*2048 dwords
    unsigned int* wpack_d = wpack1 + 2 * 72 * 2048;             // 2*72*2048
    unsigned int* wpack_s = wpack_d + 2 * 72 * 2048;            // 2*8*2048

    dim3 blk(256);

    build_wcat_kernel<<<(32 * 2304 + 255) / 256, blk, 0, stream>>>(
        w_off, b_off, w_mod, b_mod, wcat, bcat);
    prepack_w_kernel<<<(2 * 72 * 2048 + 255) / 256, blk, 0, stream>>>(w1, wpack1, 256, 9, 2);
    prepack_w_kernel<<<(2 * 72 * 2048 + 255) / 256, blk, 0, stream>>>(w_d, wpack_d, 256, 9, 2);
    prepack_w_kernel<<<(2 * 8 * 2048 + 255) / 256, blk, 0, stream>>>(w_ds, wpack_s, 256, 1, 2);

    // conv1 (3x3, 256->256) -> rawA
    conv_wmma128_kernel<3, 1><<<dim3(B_ * H_, CO_ / 128), blk, 0, stream>>>(
        x, wpack1, b1, rawA, CI_, CO_);
    bn_stats_kernel<<<CO_, blk, 0, stream>>>(rawA, mean1, var1);
    bn_relu_kernel<<<NE_ / 4 / 256, blk, 0, stream>>>(rawA, mean1, var1, g1, bt1, out1);

    // offset+mod conv (3x3, 256->32 padded) -> offmod
    conv_wmma32_kernel<3, 1><<<dim3(B_ * H_, 1), blk, 0, stream>>>(
        out1, wcat, bcat, offmod, CI_, 32);

    // 1x1 downsample (256->256) -> dsraw
    conv_wmma128_kernel<1, 0><<<dim3(B_ * H_, CO_ / 128), blk, 0, stream>>>(
        x, wpack_s, b_ds, dsraw, CI_, CO_);

    // deformable sampling prep, then deform GEMM -> rawA (reused)
    deform_prep_kernel<<<(B_ * 9 * HW_ + 255) / 256, blk, 0, stream>>>(offmod, prep);
    deform_wmma_kernel<<<dim3(B_ * H_, CO_ / 128), blk, 0, stream>>>(
        out1, prep, wpack_d, b_d, rawA);

    bn_stats_kernel<<<CO_, blk, 0, stream>>>(rawA, mean2, var2);
    bn_stats_kernel<<<CO_, blk, 0, stream>>>(dsraw, mean3, var3);

    final_fuse_kernel<<<NE_ / 4 / 256, blk, 0, stream>>>(
        rawA, mean2, var2, g2, bt2, dsraw, mean3, var3, g3, bt3, (float*)d_out);
}
